// VectorQuantizer_1408749273532
// MI455X (gfx1250) — matrix operand order repaired and checked by reading, not compiled, not run on hardware
//
#include <hip/hip_runtime.h>
#include <hip/hip_bf16.h>

// ---------------------------------------------------------------------------
// VQ-VAE vector quantizer for MI455X (gfx1250, wave32, WMMA).
//   x:        [32, 256, 32, 32] f32   -> N=32768 tokens, C=256
//   codebook: [8192, 256] f32         -> K=8192 codes
// Distance GEMM runs as f16 WMMA (v_wmma_f32_16x16x32_f16) with f32 accum;
// codebook pre-scaled by 8192 so its tiny uniform(-1/K,1/K) values use the
// full f16 mantissa. Dominant output cost is the 1 GB one-hot matrix ->
// streamed with non-temporal wide stores.
// ---------------------------------------------------------------------------

typedef _Float16 v16h __attribute__((ext_vector_type(16)));
typedef _Float16 v8h  __attribute__((ext_vector_type(8)));
typedef float    v8f  __attribute__((ext_vector_type(8)));
typedef float    v2f  __attribute__((ext_vector_type(2)));

#define BDIM   32
#define CDIM   256
#define HDIM   32
#define WDIM   32
#define NTOK   32768            // B*H*W
#define KCODE  8192
#define NC     (NTOK * CDIM)    // 8,388,608 elements of x / x_quantized
#define BETA   0.25f

// ------------------------- kernel 0: zero counts ---------------------------
__global__ void vq_zero_counts(int* __restrict__ counts) {
    int i = blockIdx.x * blockDim.x + threadIdx.x;
    if (i < KCODE) counts[i] = 0;
}

// -------------- kernel 1: transpose [B,C,H,W] -> Xh f16 [N,C] --------------
__global__ void vq_transpose_f16(const float* __restrict__ x,
                                 _Float16* __restrict__ Xh) {
    long i = (long)blockIdx.x * blockDim.x + threadIdx.x;   // over NC
    if (i >= NC) return;
    int w = (int)(i % WDIM);
    int h = (int)((i / WDIM) % HDIM);
    int c = (int)((i / (WDIM * HDIM)) % CDIM);
    int b = (int)(i / (WDIM * HDIM * CDIM));
    int n = b * (HDIM * WDIM) + h * WDIM + w;
    Xh[(long)n * CDIM + c] = (_Float16)x[i];
}

// ---- kernel 2: codebook -> f16 scaled by 8192, plus exact f32 norms² ------
__global__ __launch_bounds__(256)
void vq_codebook_prep(const float* __restrict__ cb,
                      _Float16* __restrict__ CBh,
                      float* __restrict__ enorm) {
    __shared__ float red[256];
    int k = blockIdx.x;           // one code per block
    int t = threadIdx.x;          // one channel per thread (CDIM == 256)
    float v = cb[(long)k * CDIM + t];
    CBh[(long)k * CDIM + t] = (_Float16)(v * 8192.0f);
    red[t] = v * v;
    __syncthreads();
    for (int s = 128; s > 0; s >>= 1) {
        if (t < s) red[t] += red[t + s];
        __syncthreads();
    }
    if (t == 0) enorm[k] = red[0];
}

// ------------- kernel 3: WMMA distance GEMM + per-row argmin ---------------
// Block = 256 threads = 8 waves; block owns 128 rows; wave owns 16 rows.
// Wave keeps its full 16x256 f16 A-slab in 8 fragments and sweeps all 8192
// codes in 16-column chunks (8 chained WMMAs per chunk = full C reduction).
__global__ __launch_bounds__(256)
void vq_argmin_wmma(const _Float16* __restrict__ Xh,
                    const _Float16* __restrict__ CBh,
                    const float* __restrict__ enorm,
                    int* __restrict__ idx_out) {
    const int lane = threadIdx.x & 31;
    const int wave = threadIdx.x >> 5;
    const int m0   = blockIdx.x * 128 + wave * 16;
    const int half = lane >> 4;           // lane group 0 / 1
    const int l16  = lane & 15;

    // A fragments: 16-bit A 16x32, contiguous k order (matches B):
    //   lanes 0-15: row=l16, K=0..15
    //   lanes16-31: row=l16, K=16..31
    v16h afrag[8];
    const _Float16* xrow = Xh + (long)(m0 + l16) * CDIM;
    #pragma unroll
    for (int s = 0; s < 8; ++s) {
        v8h lo = *(const v8h*)(xrow + s * 32 + half * 16);
        v8h hi = *(const v8h*)(xrow + s * 32 + half * 16 + 8);
        v16h a;
        #pragma unroll
        for (int i = 0; i < 8; ++i) { a[i] = lo[i]; a[8 + i] = hi[i]; }
        afrag[s] = a;
    }

    float minval[8];
    int   minidx[8];
    #pragma unroll
    for (int r = 0; r < 8; ++r) { minval[r] = 3.4e38f; minidx[r] = 0x7fffffff; }

    for (int chunk = 0; chunk < KCODE / 16; ++chunk) {
        const int bcol = chunk * 16 + l16;      // this lane's candidate column
        // B fragment: 16-bit B 32x16, contiguous k order (matches A): lane l -> col l%16,
        // halves h=0..15 -> K = (l/16)*16 + h  (contiguous 32B run).
        const _Float16* brow = CBh + (long)bcol * CDIM + half * 16;
        if (chunk + 1 < KCODE / 16)
            __builtin_prefetch(brow + 16 * CDIM, 0, 0);   // global_prefetch_b8

        v8f acc = {};
        #pragma unroll
        for (int s = 0; s < 8; ++s) {
            v16h b = *(const v16h*)(brow + s * 32);
            acc = __builtin_amdgcn_wmma_f32_16x16x32_f16(
                false, afrag[s], false, b, (short)0, acc, false, false);
        }
        const float en = enorm[bcol];
        #pragma unroll
        for (int r = 0; r < 8; ++r) {
            // d = ||e||^2 - 2*(x.e) ; acc holds (x . e*8192)
            float d = en - acc[r] * (2.0f / 8192.0f);
            if (d < minval[r]) { minval[r] = d; minidx[r] = bcol; }
        }
    }

    // Butterfly min-reduce across the 16 lanes that share a row group
    // (xor masks 1..8 never cross the 16-lane halves).
    #pragma unroll
    for (int m = 1; m < 16; m <<= 1) {
        #pragma unroll
        for (int r = 0; r < 8; ++r) {
            float ov = __shfl_xor(minval[r], m, 32);
            int   oi = __shfl_xor(minidx[r], m, 32);
            if (ov < minval[r] || (ov == minval[r] && oi < minidx[r])) {
                minval[r] = ov; minidx[r] = oi;
            }
        }
    }
    // C/D layout: lanes 0-15 VGPR r -> row m0+r ; lanes 16-31 -> row m0+8+r
    if (l16 == 0) {
        #pragma unroll
        for (int r = 0; r < 8; ++r)
            idx_out[m0 + half * 8 + r] = minidx[r];
    }
}

// ------- kernel 4: gather x_quantized + per-block loss partial sums --------
__global__ __launch_bounds__(256)
void vq_gather_loss(const float* __restrict__ x,
                    const float* __restrict__ cb,
                    const int* __restrict__ idx,
                    float* __restrict__ xq_out,
                    float* __restrict__ partials) {
    __shared__ float r1[256], r2[256];
    long i = (long)blockIdx.x * blockDim.x + threadIdx.x;   // over NC
    float d1 = 0.f, d2 = 0.f;
    if (i < NC) {
        int w = (int)(i % WDIM);
        int h = (int)((i / WDIM) % HDIM);
        int c = (int)((i / (WDIM * HDIM)) % CDIM);
        int b = (int)(i / (WDIM * HDIM * CDIM));
        int n = b * (HDIM * WDIM) + h * WDIM + w;
        float xv = x[i];
        float qv = cb[(long)idx[n] * CDIM + c];
        xq_out[i] = qv;                    // straight-through value == code
        float diff = xv - qv;
        d1 = diff;                         // for BETA * mean(xt - xq)
        d2 = diff * diff;                  // for mean((xt - xq)^2)
    }
    int t = threadIdx.x;
    r1[t] = d1; r2[t] = d2;
    __syncthreads();
    for (int s = 128; s > 0; s >>= 1) {
        if (t < s) { r1[t] += r1[t + s]; r2[t] += r2[t + s]; }
        __syncthreads();
    }
    if (t == 0) {
        partials[2 * blockIdx.x + 0] = r1[0];
        partials[2 * blockIdx.x + 1] = r2[0];
    }
}

// ------------------- kernel 5: histogram of indices ------------------------
__global__ void vq_histogram(const int* __restrict__ idx,
                             int* __restrict__ counts) {
    int n = blockIdx.x * blockDim.x + threadIdx.x;
    if (n < NTOK) atomicAdd(&counts[idx[n]], 1);
}

// --------- kernel 6: finalize loss + perplexity (single block) -------------
__global__ __launch_bounds__(256)
void vq_finalize(const float* __restrict__ partials, int nPart,
                 const int* __restrict__ counts,
                 float* __restrict__ loss_out,
                 float* __restrict__ perp_out) {
    __shared__ float r1[256], r2[256], r3[256];
    int t = threadIdx.x;
    float s1 = 0.f, s2 = 0.f, sH = 0.f;
    for (int j = t; j < nPart; j += 256) {
        s1 += partials[2 * j + 0];
        s2 += partials[2 * j + 1];
    }
    for (int k = t; k < KCODE; k += 256) {
        float em = (float)counts[k] / (float)NTOK;
        sH += em * logf(em + 1e-10f);
    }
    r1[t] = s1; r2[t] = s2; r3[t] = sH;
    __syncthreads();
    for (int s = 128; s > 0; s >>= 1) {
        if (t < s) { r1[t] += r1[t + s]; r2[t] += r2[t + s]; r3[t] += r3[t + s]; }
        __syncthreads();
    }
    if (t == 0) {
        float invNC = 1.0f / (float)NC;
        *loss_out = BETA * (r1[0] * invNC) + (r2[0] * invNC);
        *perp_out = expf(-r3[0]);
    }
}

// -------- kernel 7: stream-zero the 1 GB one-hot region (NT stores) --------
// Region starts at float offset NC+2 (byte offset % 16 == 8), so 8-byte
// ext-vector stores are the widest aligned option.
__global__ void vq_fill_zero2(float* __restrict__ p, long n2) {
    long i = (long)blockIdx.x * blockDim.x + threadIdx.x;
    if (i < n2) {
        v2f z = {0.f, 0.f};
        __builtin_nontemporal_store(z, (v2f*)(p + 2 * i));
    }
}

// ------ kernel 8: scatter ones into one-hot + emit indices as float --------
__global__ void vq_scatter(const int* __restrict__ idx,
                           float* __restrict__ enc,
                           float* __restrict__ idx_f) {
    int n = blockIdx.x * blockDim.x + threadIdx.x;
    if (n < NTOK) {
        enc[(long)n * KCODE + idx[n]] = 1.0f;
        idx_f[n] = (float)idx[n];
    }
}

// ---------------------------------------------------------------------------
extern "C" void kernel_launch(void* const* d_in, const int* in_sizes, int n_in,
                              void* d_out, int out_size, void* d_ws, size_t ws_size,
                              hipStream_t stream) {
    const float* x  = (const float*)d_in[0];   // [32,256,32,32]
    const float* cb = (const float*)d_in[1];   // [8192,256]
    float* out = (float*)d_out;

    // output layout (flat f32 concat, reference return order)
    float* out_xq   = out;                          // NC
    float* out_loss = out + (long)NC;               // 1
    float* out_perp = out + (long)NC + 1;           // 1
    float* out_enc  = out + (long)NC + 2;           // NTOK*KCODE
    float* out_idxf = out_enc + (long)NTOK * KCODE; // NTOK

    // workspace layout
    char* ws = (char*)d_ws;
    _Float16* Xh     = (_Float16*)(ws);                        // 16 MB
    _Float16* CBh    = (_Float16*)(ws + (16u << 20));          //  4 MB
    float*    enorm  = (float*)   (ws + (20u << 20));          // 32 KB
    int*      idx    = (int*)     (ws + (20u << 20) + (64u << 10)); // 128 KB
    int*      counts = (int*)     (ws + (20u << 20) + (256u << 10));// 32 KB
    float*    part   = (float*)   (ws + (20u << 20) + (512u << 10));// 256 KB

    const int nLossBlocks = NC / 256;                 // 32768

    vq_zero_counts   <<<KCODE / 256, 256, 0, stream>>>(counts);
    vq_transpose_f16 <<<NC / 256, 256, 0, stream>>>(x, Xh);
    vq_codebook_prep <<<KCODE, 256, 0, stream>>>(cb, CBh, enorm);
    vq_argmin_wmma   <<<NTOK / 128, 256, 0, stream>>>(Xh, CBh, enorm, idx);
    vq_gather_loss   <<<nLossBlocks, 256, 0, stream>>>(x, cb, idx, out_xq, part);
    vq_histogram     <<<NTOK / 256, 256, 0, stream>>>(idx, counts);
    vq_finalize      <<<1, 256, 0, stream>>>(part, nLossBlocks, counts,
                                             out_loss, out_perp);
    const long n2 = ((long)NTOK * KCODE) / 2;         // float2 count
    vq_fill_zero2    <<<(unsigned)(n2 / 256), 256, 0, stream>>>(out_enc, n2);
    vq_scatter       <<<NTOK / 256, 256, 0, stream>>>(idx, out_enc, out_idxf);
}
